// GenLSTMd_9938554323550
// MI455X (gfx1250) — matrix operand order repaired and checked by reading, not compiled, not run on hardware
//
#include <hip/hip_runtime.h>

typedef __attribute__((ext_vector_type(16))) _Float16 v16h;
typedef __attribute__((ext_vector_type(8)))  _Float16 v8h;
typedef __attribute__((ext_vector_type(8)))  float    v8f;

#define COND_STEPS 31
#define TOTAL_STEPS 255            // 31 conditioning + 224 generation
#define HS 272                     // h row stride in halves (544 B, 16B aligned)
#define IS 48                      // input row stride in halves (96 B, 16B aligned)

#define WR_HALVES (64*8*32*16)     // 64 n-tiles x 8 k-steps x 32 lanes x 16 halves
#define WK_HALVES (64*32*16)       // 64 n-tiles x 1 k-step  x 32 lanes x 16 halves
#define WLDS_SEGS (64*4*32)        // n-tiles x k-steps 4..7 x lanes (16 halves each)

// ---------------------------------------------------------------------------
// Prep: swizzle f32 weights into per-lane WMMA B-fragment layout (f16) in ws.
// B 32x16 layout: lane L (<16): N=L, halves = K 0..7,16..23 ; lane L+16: N=L, K 8..15,24..31
// Bias is baked into packed-Wk row K=18 (input feature 18 is constant 1.0),
// so accumulators start from the WMMA inline-zero C operand.
// ---------------------------------------------------------------------------
__global__ void pack_weights_kernel(const float* __restrict__ Wk,
                                    const float* __restrict__ Wr,
                                    const float* __restrict__ b_lstm,
                                    _Float16* __restrict__ wrp,
                                    _Float16* __restrict__ wkp) {
  int id = blockIdx.x * blockDim.x + threadIdx.x;
  if (id < WR_HALVES) {
    int j = id & 15, lane = (id >> 4) & 31, ks = (id >> 9) & 7, nt = id >> 12;
    int N = nt * 16 + (lane & 15);
    int K = ks * 32 + ((j < 8) ? 0 : 8) + ((lane >= 16) ? 8 : 0) + j;
    wrp[id] = (_Float16)Wr[K * 1024 + N];
  } else {
    id -= WR_HALVES;
    if (id < WK_HALVES) {
      int j = id & 15, lane = (id >> 4) & 31, nt = id >> 9;
      int N = nt * 16 + (lane & 15);
      int K = ((j < 8) ? 0 : 8) + ((lane >= 16) ? 8 : 0) + j;
      float v = (K < 18) ? Wk[K * 1024 + N] : ((K == 18) ? b_lstm[N] : 0.f);
      wkp[id] = (_Float16)v;
    }
  }
}

__device__ __forceinline__ float sig_(float x) { return 1.f / (1.f + __expf(-x)); }
__device__ __forceinline__ float tanh_(float x) {
  float z = fminf(fmaxf(2.f * x, -30.f), 30.f);
  float e = __expf(z);
  return (e - 1.f) / (e + 1.f);
}

union AFrag { v16h v; v8h h[2]; };

// ---------------------------------------------------------------------------
// Persistent-weight LSTM: 64 WGs x 512 thr (16 waves). WG = 16 batch rows.
// Wave w owns hidden cols [16w,16w+16) for gates i/f/g/o (z tiles w,w+16,w+32,w+48).
// Wr K=0..127: 128 VGPRs/wave. Wr K=128..255: 256KB LDS (CDNA5 WGP has 320KB),
// pre-swizzled to B-fragment form. Wk+bias: streamed from L2 every step.
// ---------------------------------------------------------------------------
__launch_bounds__(512)
__global__ void genlstm_kernel(const float* __restrict__ noise,   // (1024,255,16)
                               const float* __restrict__ x,       // (1024,256,2)
                               const float* __restrict__ Wd,      // (256,)
                               const float* __restrict__ bd,      // (1,)
                               const _Float16* __restrict__ wrp,
                               const _Float16* __restrict__ wkp,
                               float* __restrict__ out) {         // (1024,256,2)
  __shared__ _Float16 wlds[WLDS_SEGS * 16];  // 256 KB: Wr fragments, k-steps 4..7
  __shared__ _Float16 hbuf[16 * HS];         // h state, f16
  __shared__ _Float16 inp[16 * IS];          // step input, K-padded to 32
  __shared__ float    WdL[256];
  __shared__ float    xgL[16];

  const int tid  = threadIdx.x;
  const int lane = tid & 31;
  const int wid  = tid >> 5;          // wave id 0..15 == hidden-column tile
  const int b0   = blockIdx.x * 16;   // first batch row of this WG

  // ---- persistent weights: K=0..127 into VGPRs (one-time L2 read) ----
  v16h wr[4][4];
  #pragma unroll
  for (int j = 0; j < 4; ++j) {
    const int tile = wid + 16 * j;    // j: 0=i 1=f 2=g 3=o
    #pragma unroll
    for (int ks = 0; ks < 4; ++ks)
      wr[j][ks] = *(const v16h*)(wrp + (((tile * 8 + ks) * 32 + lane) << 4));
  }

  // ---- weights K=128..255 into LDS (one-time, stays for whole sequence) ----
  for (int s = tid; s < WLDS_SEGS; s += 512) {
    int lane_s = s & 31, ks4 = (s >> 5) & 3, tile = s >> 7;
    *(v16h*)(wlds + (s << 4)) =
        *(const v16h*)(wrp + (((tile * 8 + 4 + ks4) * 32 + lane_s) << 4));
  }

  // ---- init LDS / outputs ----
  for (int i = tid; i < 16 * HS; i += 512) hbuf[i] = (_Float16)0.f;
  for (int i = tid; i < 16 * IS; i += 512)       // feature 18 == constant 1 (bias)
    inp[i] = (_Float16)(((i % IS) == 18) ? 1.f : 0.f);
  if (tid < 256) WdL[tid] = Wd[tid];
  for (int i = tid; i < 16 * 256; i += 512) {   // t column passthrough
    int row = i >> 8, s = i & 255;
    out[((size_t)(b0 + row) * 256 + s) * 2] = x[((size_t)(b0 + row) * 256 + s) * 2];
  }
  float cum = 0.f;
  if (tid < 16) out[((size_t)(b0 + tid) * 256) * 2 + 1] = 0.f;
  const float bd0 = bd[0];

  v8f c = {};   // cell-state tile (16x16 f32) for this wave's hidden cols

  #pragma unroll 1
  for (int step = 0; step < TOTAL_STEPS; ++step) {
    const bool gen = (step >= COND_STEPS);

    if (gen) {  // xg = h @ Wd + bd   (block-wide reduction via LDS float atomics)
      if (tid < 16) xgL[tid] = bd0;
      __syncthreads();
      {
        int row = tid >> 5, seg = tid & 31;
        v8h hv = *(const v8h*)(hbuf + row * HS + seg * 8);
        float s = 0.f;
        #pragma unroll
        for (int t = 0; t < 8; ++t) s += (float)hv[t] * WdL[seg * 8 + t];
        atomicAdd(&xgL[row], s);
      }
      __syncthreads();
    }

    // ---- build step input [x0, noise(16), dt] (k=18 stays 1.0 for bias) ----
    {
      int row = tid >> 5, k = tid & 31;
      size_t b = (size_t)(b0 + row);
      if (k < 18) {
        float v;
        if (k == 0)
          v = gen ? xgL[row]
                  : (x[(b * 256 + step + 1) * 2 + 1] - x[(b * 256 + step) * 2 + 1]);
        else if (k < 17)
          v = noise[(b * 255 + step) * 16 + (k - 1)];
        else
          v = x[(b * 256 + step + 1) * 2] - x[(b * 256 + step) * 2];
        inp[row * IS + k] = (_Float16)v;
      }
    }
    __syncthreads();   // orders prev h writes + wlds init + inp build before reads

    // ---- running cumsum output column ----
    if (tid < 16) {
      size_t b = (size_t)(b0 + tid);
      float inc = gen ? xgL[tid]
                      : (x[(b * 256 + step + 1) * 2 + 1] - x[(b * 256 + step) * 2 + 1]);
      cum += inc;
      out[(b * 256 + step + 1) * 2 + 1] = cum;
    }

    // ---- z = inp@Wk' + h@Wr  via WMMA (bias baked into Wk' row 18) ----
    v8f acc[4] = {};
    {
      const int row  = lane & 15;
      const int koff = (lane >= 16) ? 8 : 0;   // A 16x32 layout per ISA

      // Wk streamed from L2 each step; opaque offset blocks LICM from
      // hoisting these loads into (scarce) persistent VGPRs.
      long zo = 0;
      asm volatile("" : "+s"(zo));
      const _Float16* wkq = wkp + zo;

      v16h wkf[4];
      #pragma unroll
      for (int j = 0; j < 4; ++j)
        wkf[j] = *(const v16h*)(wkq + ((((wid + 16 * j) * 32) + lane) << 4));
      AFrag a;
      a.h[0] = *(const v8h*)(inp + row * IS + koff);
      a.h[1] = *(const v8h*)(inp + row * IS + koff + 16);
      #pragma unroll
      for (int j = 0; j < 4; ++j)
        acc[j] = __builtin_amdgcn_wmma_f32_16x16x32_f16(false, a.v, false, wkf[j],
                                                        (short)0, acc[j], false, false);

      // recurrent K = 0..127 : B fragments resident in VGPRs
      #pragma unroll
      for (int ks = 0; ks < 4; ++ks) {
        AFrag ah;
        ah.h[0] = *(const v8h*)(hbuf + row * HS + ks * 32 + koff);
        ah.h[1] = *(const v8h*)(hbuf + row * HS + ks * 32 + koff + 16);
        #pragma unroll
        for (int j = 0; j < 4; ++j)
          acc[j] = __builtin_amdgcn_wmma_f32_16x16x32_f16(false, ah.v, false, wr[j][ks],
                                                          (short)0, acc[j], false, false);
      }
      // recurrent K = 128..255 : B fragments from LDS, 4 gates batched
      #pragma unroll
      for (int ks4 = 0; ks4 < 4; ++ks4) {
        AFrag ah;
        ah.h[0] = *(const v8h*)(hbuf + row * HS + (4 + ks4) * 32 + koff);
        ah.h[1] = *(const v8h*)(hbuf + row * HS + (4 + ks4) * 32 + koff + 16);
        v16h wbf[4];
        #pragma unroll
        for (int j = 0; j < 4; ++j)
          wbf[j] = *(const v16h*)(wlds + ((((wid + 16 * j) * 4 + ks4) * 32 + lane) << 4));
        #pragma unroll
        for (int j = 0; j < 4; ++j)
          acc[j] = __builtin_amdgcn_wmma_f32_16x16x32_f16(false, ah.v, false, wbf[j],
                                                          (short)0, acc[j], false, false);
      }
    }
    __syncthreads();   // all hbuf reads complete before rewrite

    // ---- gates + state update; write h back to LDS as f16 ----
    {
      const int colb = wid * 16 + (lane & 15);
      const int rb   = (lane >= 16) ? 8 : 0;   // C/D layout: vgpr r -> rows r, r+8
      #pragma unroll
      for (int r = 0; r < 8; ++r) {
        float iv = sig_(acc[0][r]);
        float fv = sig_(acc[1][r]);
        float gv = tanh_(acc[2][r]);
        float ov = sig_(acc[3][r]);
        c[r] = fv * c[r] + iv * gv;
        float hv = ov * tanh_(c[r]);
        hbuf[(r + rb) * HS + colb] = (_Float16)hv;
      }
    }
  }
}

extern "C" void kernel_launch(void* const* d_in, const int* in_sizes, int n_in,
                              void* d_out, int out_size, void* d_ws, size_t ws_size,
                              hipStream_t stream) {
  const float* noise  = (const float*)d_in[0];
  const float* x      = (const float*)d_in[1];
  const float* Wk     = (const float*)d_in[2];
  const float* Wr     = (const float*)d_in[3];
  const float* b_lstm = (const float*)d_in[4];
  const float* Wd     = (const float*)d_in[5];
  const float* bd     = (const float*)d_in[6];
  float* out = (float*)d_out;

  _Float16* wrp = (_Float16*)d_ws;
  _Float16* wkp = wrp + WR_HALVES;

  const int total = WR_HALVES + WK_HALVES;
  pack_weights_kernel<<<(total + 255) / 256, 256, 0, stream>>>(Wk, Wr, b_lstm, wrp, wkp);
  genlstm_kernel<<<64, 512, 0, stream>>>(noise, x, Wd, bd, wrp, wkp, out);
}
